// YOLOF_12077448036524
// MI455X (gfx1250) — matrix operand order repaired and checked by reading, not compiled, not run on hardware
//
#include <hip/hip_runtime.h>
#include <hip/hip_bf16.h>
#include <math.h>

typedef __attribute__((ext_vector_type(16))) __bf16 v16bf;
typedef __attribute__((ext_vector_type(8)))  float  v8f;

#define HH 80
#define WW 80
#define CIN 512
#define NPX 6400
#define NA 5
#define NC 80
#define COUT_PAD 448      // 25 cls tiles (400) + 2 reg tiles (32) + 1 obj tile (16)
#define NTILE 28
#define CLS_TILES 25
#define NSEL 1000
#define NSCORE 32000
#define NSTEP 72          // 9 taps * 8 chunks of 64 channels (even -> 2-step unroll)
#define SCALE_CLAMP_F 4.135166556742356f
#define CTR_CLAMP_F 32.0f
#define CONF_THRESH_F 0.05f
#define NMS_THRESH_F 0.6f

__device__ __forceinline__ float sigmoidf_(float x) { return 1.0f / (1.0f + expf(-x)); }

__device__ __forceinline__ unsigned short f2bf(float x) {
    unsigned u = __float_as_uint(x);
    unsigned r = (u + 0x7FFFu + ((u >> 16) & 1u)) >> 16;
    return (unsigned short)r;
}

// ---------------- feature convert: [512][80][80] f32 -> [6400][512] bf16 (NHWC) ------------
__global__ void cvt_feat_kernel(const float* __restrict__ in, unsigned short* __restrict__ out) {
    int o = blockIdx.x * blockDim.x + threadIdx.x;
    if (o >= CIN * NPX) return;
    int p = o >> 9;        // pixel
    int c = o & 511;       // channel
    out[o] = f2bf(in[c * NPX + p]);
}

// ---------------- weight pack into WMMA A-fragment layout ------------------------------------
// wpack[tile][tap][kc][lane][16] bf16 (kc = 32-channel chunk)
// lane 0..15:  M = lane,    elem e -> K = (e<8 ? e    : e+8)   (K 0..7, 16..23)
// lane 16..31: M = lane-16, elem e -> K = (e<8 ? e+8 : e+16)   (K 8..15, 24..31)
__global__ void pack_w_kernel(const float* __restrict__ w_cls,
                              const float* __restrict__ w_reg,
                              const float* __restrict__ w_obj,
                              unsigned short* __restrict__ wpack) {
    int id = blockIdx.x * blockDim.x + threadIdx.x;
    const int TOT = NTILE * 9 * 16 * 32 * 16;
    if (id >= TOT) return;
    int e = id & 15;
    int rest = id >> 4;
    int lane = rest & 31; rest >>= 5;
    int kc = rest & 15;   rest >>= 4;
    int t = rest % 9;
    int tile = rest / 9;
    int lane16 = lane & 15;
    int laneHi = lane >> 4;
    int kk = (laneHi == 0) ? (e < 8 ? e : e + 8) : (e < 8 ? e + 8 : e + 16);
    int c = kc * 32 + kk;
    int co = tile * 16 + lane16;
    float v = 0.0f;
    if (co < 400)                   v = w_cls[((size_t)co * CIN + c) * 9 + t];
    else if (co < 420)              v = w_reg[((size_t)(co - 400) * CIN + c) * 9 + t];
    else if (co >= 432 && co < 437) v = w_obj[((size_t)(co - 432) * CIN + c) * 9 + t];
    wpack[id] = f2bf(v);
}

__global__ void pack_bias_kernel(const float* __restrict__ b_cls,
                                 const float* __restrict__ b_reg,
                                 const float* __restrict__ b_obj,
                                 float* __restrict__ biasp) {
    int i = blockIdx.x * blockDim.x + threadIdx.x;
    if (i >= COUT_PAD) return;
    float v = 0.0f;
    if (i < 400)                   v = b_cls[i];
    else if (i < 420)              v = b_reg[i - 400];
    else if (i >= 432 && i < 437)  v = b_obj[i - 432];
    biasp[i] = v;
}

// ---------------- implicit-GEMM 3x3 conv via bf16 WMMA with shared LDS B-stage ---------------
// One block = several waves, each owning one 16-channel co-tile, all sharing the SAME
// 64-pixel group and the SAME feature map. Per step (tap, 64-channel chunk):
//   * cooperative global->reg prefetch of the next 8KB B tile (statically double-buffered LDS)
//   * each wave: 2 A fragment loads + 8 LDS B fragment loads + 8 wmma
//   * one barrier per step; 2-step unroll keeps LDS bases static
__global__ __launch_bounds__(256) void conv_wmma_lds_kernel(
    const unsigned short* __restrict__ feat,      // [6400][512] bf16
    const unsigned short* __restrict__ wpack,     // [28][9][16][32][16] bf16
    const float* __restrict__ biasp,              // [448]
    float* __restrict__ out,                      // [6400][448] f32
    int tileBase)
{
    __shared__ unsigned short stage[2][4096];     // 2 x 8KB: [kcH(2)][u(4)][lane(32)][16]
    const int tid  = threadIdx.x;
    const int nthr = blockDim.x;
    const int lane = tid & 31;
    const int wid  = tid >> 5;
    const int pg   = blockIdx.x % 100;
    const int cog  = blockIdx.x / 100;
    const int tile = tileBase + cog * (nthr >> 5) + wid;
    const int px_base = pg * 64;
    const int lane16 = lane & 15;
    const int laneHi = lane >> 4;

    int yy[4], xb[4];
#pragma unroll
    for (int u = 0; u < 4; ++u) {
        int pb = px_base + u * 16;
        yy[u] = pb / WW;
        xb[u] = pb % WW;
    }
    const int DY[9] = { -1, -1, -1, 0, 0, 0, 1, 1, 1 };
    const int DX[9] = { -1, 0, 1, -1, 0, 1, -1, 0, 1 };

    v8f acc0 = {}, acc1 = {}, acc2 = {}, acc3 = {};

    // chunk c (16 bytes) of a step's 8KB B tile -> global source (zero-filled at borders)
    auto loadChunk = [&](int step, int c) -> uint4 {
        int t   = step >> 3;
        int kc2 = step & 7;           // 64-channel chunk
        int half = c & 1;
        int slot = c >> 1;            // 0..255
        int kcH  = slot >> 7;         // 0..1  (32-channel half)
        int rem  = slot & 127;
        int u    = rem >> 5;
        int ln   = rem & 31;
        int l16  = ln & 15;
        int lHi  = ln >> 4;
        int sy = yy[u] + DY[t];
        int sx = xb[u] + l16 + DX[t];
        uint4 v = make_uint4(0u, 0u, 0u, 0u);
        if ((unsigned)sy < (unsigned)HH && (unsigned)sx < (unsigned)WW) {
            v = *(const uint4*)(feat + ((size_t)(sy * WW + sx) * CIN + kc2 * 64 + kcH * 32 + lHi * 8 + half * 16));
        }
        return v;
    };

    // compute one step (two 32-channel halves, 8 wmma) from a fixed LDS buffer
    auto computeStep = [&](const unsigned short* sbuf, int step) {
        int t   = step >> 3;
        int kc2 = step & 7;
        const unsigned short* ap = wpack + ((size_t)((tile * 9 + t) * 16 + kc2 * 2) * 32 + lane) * 16;
#pragma unroll
        for (int h = 0; h < 2; ++h) {
            v16bf a = *(const v16bf*)(ap + h * 512);
            const unsigned short* sb = sbuf + h * 2048 + lane * 16;
            v16bf b0 = *(const v16bf*)(sb + 0 * 512);
            acc0 = __builtin_amdgcn_wmma_f32_16x16x32_bf16(false, a, false, b0, (short)0, acc0, false, false);
            v16bf b1 = *(const v16bf*)(sb + 1 * 512);
            acc1 = __builtin_amdgcn_wmma_f32_16x16x32_bf16(false, a, false, b1, (short)0, acc1, false, false);
            v16bf b2 = *(const v16bf*)(sb + 2 * 512);
            acc2 = __builtin_amdgcn_wmma_f32_16x16x32_bf16(false, a, false, b2, (short)0, acc2, false, false);
            v16bf b3 = *(const v16bf*)(sb + 3 * 512);
            acc3 = __builtin_amdgcn_wmma_f32_16x16x32_bf16(false, a, false, b3, (short)0, acc3, false, false);
        }
        __builtin_prefetch((const void*)(ap + 1024), 0, 0);   // next A fragment stream
    };

    // prime buffer 0 with step 0
    for (int c = tid; c < 512; c += nthr) {
        uint4 v = loadChunk(0, c);
        *(uint4*)(&stage[0][c * 8]) = v;
    }
    __syncthreads();

#pragma unroll 1
    for (int step = 0; step < NSTEP; step += 2) {
        // phase A: prefetch step+1, compute step from buf0, commit to buf1
        {
            uint4 pv[6];
            int npf = 0;
            for (int c = tid; c < 512; c += nthr) pv[npf++] = loadChunk(step + 1, c);
            computeStep(&stage[0][0], step);
            int i = 0;
            for (int c = tid; c < 512; c += nthr) { *(uint4*)(&stage[1][c * 8]) = pv[i++]; }
        }
        __syncthreads();
        // phase B: prefetch step+2 (if any), compute step+1 from buf1, commit to buf0
        {
            const bool more = (step + 2) < NSTEP;
            uint4 pv[6];
            int npf = 0;
            if (more) { for (int c = tid; c < 512; c += nthr) pv[npf++] = loadChunk(step + 2, c); }
            computeStep(&stage[1][0], step + 1);
            if (more) { int i = 0; for (int c = tid; c < 512; c += nthr) { *(uint4*)(&stage[0][c * 8]) = pv[i++]; } }
        }
        __syncthreads();
    }

    // bias add: component r -> co = tile*16 + laneHi*8 + r
    const float* bp = biasp + tile * 16 + laneHi * 8;
    float bias[8];
#pragma unroll
    for (int r = 0; r < 8; ++r) bias[r] = bp[r];
#pragma unroll
    for (int r = 0; r < 8; ++r) { acc0[r] += bias[r]; acc1[r] += bias[r]; acc2[r] += bias[r]; acc3[r] += bias[r]; }

    const int cobase = tile * 16 + laneHi * 8;
#pragma unroll
    for (int u = 0; u < 4; ++u) {
        int p = px_base + u * 16 + lane16;
        float* op = out + (size_t)p * COUT_PAD + cobase;
        v8f acc = (u == 0) ? acc0 : (u == 1) ? acc1 : (u == 2) ? acc2 : acc3;
        *(float4*)(op)     = make_float4(acc[0], acc[1], acc[2], acc[3]);
        *(float4*)(op + 4) = make_float4(acc[4], acc[5], acc[6], acc[7]);
    }
}

// ---------------- per (pixel, anchor) score/label --------------------------------------------
__global__ void score_kernel(const float* __restrict__ conv,
                             float* __restrict__ score, int* __restrict__ label) {
    int i = blockIdx.x * blockDim.x + threadIdx.x;
    if (i >= NSCORE) return;
    int p = i / NA, a = i % NA;
    const float* base = conv + (size_t)p * COUT_PAD + a * NC;
    float mx = -__builtin_inff();
    int am = 0;
#pragma unroll 8
    for (int c = 0; c < NC; ++c) {
        float v = base[c];
        if (v > mx) { mx = v; am = c; }
    }
    float obj = conv[(size_t)p * COUT_PAD + 432 + a];
    float s1 = sigmoidf_(mx) * sigmoidf_(obj);
    score[i] = sigmoidf_(s1);   // reference applies sigmoid twice
    label[i] = am;
}

// ---------------- iterative top-1000 (ties -> lowest index, matches lax.top_k) ---------------
__global__ __launch_bounds__(1024) void topk_kernel(const float* __restrict__ score,
                                                    float* __restrict__ work,
                                                    int* __restrict__ topk_idx,
                                                    float* __restrict__ topk_score) {
    const int tid = threadIdx.x;
    const int lane = tid & 31;
    const int wid = tid >> 5;
    __shared__ float redv[32];
    __shared__ int   redi[32];
    for (int i = tid; i < NSCORE; i += 1024) work[i] = score[i];
    __syncthreads();
    for (int sel = 0; sel < NSEL; ++sel) {
        float bv = -__builtin_inff();
        int bi = NSCORE;
        for (int i = tid; i < NSCORE; i += 1024) {
            float v = work[i];
            if (v > bv || (v == bv && i < bi)) { bv = v; bi = i; }
        }
#pragma unroll
        for (int off = 16; off > 0; off >>= 1) {
            float ov = __shfl_down(bv, off);
            int   oi = __shfl_down(bi, off);
            if (ov > bv || (ov == bv && oi < bi)) { bv = ov; bi = oi; }
        }
        if (lane == 0) { redv[wid] = bv; redi[wid] = bi; }
        __syncthreads();
        if (wid == 0) {
            bv = redv[lane]; bi = redi[lane];
#pragma unroll
            for (int off = 16; off > 0; off >>= 1) {
                float ov = __shfl_down(bv, off);
                int   oi = __shfl_down(bi, off);
                if (ov > bv || (ov == bv && oi < bi)) { bv = ov; bi = oi; }
            }
            if (lane == 0) {
                topk_idx[sel] = bi;
                topk_score[sel] = bv;
                work[bi] = -__builtin_inff();
            }
        }
        __syncthreads();
    }
}

// ---------------- decode + greedy class-aware NMS + output writes ----------------------------
__global__ __launch_bounds__(1024) void nms_out_kernel(const float* __restrict__ conv,
                                                       const int* __restrict__ topk_idx,
                                                       const float* __restrict__ topk_score,
                                                       const int* __restrict__ label,
                                                       const int* __restrict__ img_size_p,
                                                       float* __restrict__ out) {
    __shared__ float x1s[NSEL], y1s[NSEL], x2s[NSEL], y2s[NSEL], ars[NSEL];
    __shared__ int   lbs[NSEL];
    __shared__ int   sup[NSEL];
    const int tid = threadIdx.x;
    const float inv = 1.0f / (float)img_size_p[0];

    for (int j = tid; j < NSEL; j += 1024) {
        int idx = topk_idx[j];
        int p = idx / NA, a = idx % NA;
        float ax = ((float)(p % WW) + 0.5f) * 32.0f;
        float ay = ((float)(p / WW) + 0.5f) * 32.0f;
        float aw = (float)(32 << a);
        float ah = aw;
        const float* r = conv + (size_t)p * COUT_PAD + 400 + a * 4;
        float ox = fminf(fmaxf(r[0] * aw, -CTR_CLAMP_F), CTR_CLAMP_F);
        float oy = fminf(fmaxf(r[1] * ah, -CTR_CLAMP_F), CTR_CLAMP_F);
        float cx = ax + ox, cy = ay + oy;
        float w = aw * expf(fminf(r[2], SCALE_CLAMP_F));
        float h = ah * expf(fminf(r[3], SCALE_CLAMP_F));
        float x1 = cx - 0.5f * w, y1 = cy - 0.5f * h;
        float x2 = cx + 0.5f * w, y2 = cy + 0.5f * h;
        x1s[j] = x1; y1s[j] = y1; x2s[j] = x2; y2s[j] = y2;
        ars[j] = (x2 - x1) * (y2 - y1);
        lbs[j] = label[idx];
        float sc = topk_score[j];
        sup[j] = (sc >= CONF_THRESH_F) ? 0 : 1;
        // outputs (topk order == sort order: scores already descending)
        out[j * 4 + 0] = fminf(fmaxf(x1 * inv, 0.0f), 1.0f);
        out[j * 4 + 1] = fminf(fmaxf(y1 * inv, 0.0f), 1.0f);
        out[j * 4 + 2] = fminf(fmaxf(x2 * inv, 0.0f), 1.0f);
        out[j * 4 + 3] = fminf(fmaxf(y2 * inv, 0.0f), 1.0f);
        out[4000 + j] = sc;
        out[5000 + j] = (float)lbs[j];
    }
    __syncthreads();

    for (int i = 0; i < NSEL; ++i) {
        if (sup[i] == 0) {
            float ix1 = x1s[i], iy1 = y1s[i], ix2 = x2s[i], iy2 = y2s[i], iar = ars[i];
            int il = lbs[i];
            for (int j = tid; j < NSEL; j += 1024) {
                if (j > i && lbs[j] == il) {
                    float xx1 = fmaxf(ix1, x1s[j]);
                    float yy1 = fmaxf(iy1, y1s[j]);
                    float xx2 = fminf(ix2, x2s[j]);
                    float yy2 = fminf(iy2, y2s[j]);
                    float inter = fmaxf(1e-10f, xx2 - xx1) * fmaxf(1e-10f, yy2 - yy1);
                    float iou = inter / (iar + ars[j] - inter + 1e-14f);
                    if (iou > NMS_THRESH_F) sup[j] = 1;
                }
            }
        }
        __syncthreads();
    }

    for (int j = tid; j < NSEL; j += 1024) out[6000 + j] = sup[j] ? 0.0f : 1.0f;
}

// ---------------- launch -----------------------------------------------------------------
extern "C" void kernel_launch(void* const* d_in, const int* in_sizes, int n_in,
                              void* d_out, int out_size, void* d_ws, size_t ws_size,
                              hipStream_t stream) {
    const float* cls_feats = (const float*)d_in[0];
    const float* reg_feats = (const float*)d_in[1];
    const float* w_obj = (const float*)d_in[2];
    const float* b_obj = (const float*)d_in[3];
    const float* w_cls = (const float*)d_in[4];
    const float* b_cls = (const float*)d_in[5];
    const float* w_reg = (const float*)d_in[6];
    const float* b_reg = (const float*)d_in[7];
    const int*   img   = (const int*)d_in[8];
    float* out = (float*)d_out;

    char* ws = (char*)d_ws;
    unsigned short* featCls = (unsigned short*)(ws + 0);           // 6,553,600 B
    unsigned short* featReg = (unsigned short*)(ws + 6553600);     // 6,553,600 B
    unsigned short* wpack   = (unsigned short*)(ws + 13107200);    // 4,128,768 B
    float* biasp            = (float*)(ws + 17236224);             // 1,792 B
    float* convout          = (float*)(ws + 17238016);             // 11,468,800 B
    float* score            = (float*)(ws + 28706816);             // 128,000 B
    int*   label            = (int*)(ws + 28834816);               // 128,000 B
    float* work             = (float*)(ws + 28962816);             // 128,000 B
    int*   topk_idx         = (int*)(ws + 29090816);               // 4,000 B
    float* topk_score       = (float*)(ws + 29094912);             // 4,000 B

    // 1) convert features to bf16 NHWC
    {
        int n = CIN * NPX;
        int blocks = (n + 255) / 256;
        cvt_feat_kernel<<<blocks, 256, 0, stream>>>(cls_feats, featCls);
        cvt_feat_kernel<<<blocks, 256, 0, stream>>>(reg_feats, featReg);
    }
    // 2) pack weights into WMMA A-fragment layout
    {
        int n = NTILE * 9 * 16 * 32 * 16;
        pack_w_kernel<<<(n + 255) / 256, 256, 0, stream>>>(w_cls, w_reg, w_obj, wpack);
        pack_bias_kernel<<<2, 256, 0, stream>>>(b_cls, b_reg, b_obj, biasp);
    }
    // 3) fused conv via WMMA + shared LDS B-stage:
    //    cls:      25 tiles = 5 groups x 5 waves/block, 100 pixel-groups -> 500 blocks x 160 thr
    //    reg+obj:   3 tiles = 1 group  x 3 waves/block, 100 pixel-groups -> 100 blocks x  96 thr
    conv_wmma_lds_kernel<<<500, 160, 0, stream>>>(featCls, wpack, biasp, convout, 0);
    conv_wmma_lds_kernel<<<100,  96, 0, stream>>>(featReg, wpack, biasp, convout, 25);
    // 4) sigmoid(sigmoid(max_cls)*sigmoid(obj)) scores + argmax labels
    score_kernel<<<(NSCORE + 255) / 256, 256, 0, stream>>>(convout, score, label);
    // 5) top-1000
    topk_kernel<<<1, 1024, 0, stream>>>(score, work, topk_idx, topk_score);
    // 6) decode + NMS + outputs
    nms_out_kernel<<<1, 1024, 0, stream>>>(convout, topk_idx, topk_score, label, img, out);
}
